// SelfAttention_58772332479046
// MI455X (gfx1250) — matrix operand order repaired and checked by reading, not compiled, not run on hardware
//
#include <hip/hip_runtime.h>
#include <hip/hip_bf16.h>

typedef __attribute__((ext_vector_type(16))) __bf16 v16bf;
typedef __attribute__((ext_vector_type(8)))  __bf16 bf16x8;
typedef __attribute__((ext_vector_type(4)))  __bf16 bf16x4;
typedef __attribute__((ext_vector_type(8)))  float  v8f;
typedef __attribute__((ext_vector_type(4)))  unsigned int u32x4;
typedef __attribute__((ext_vector_type(8)))  int i32x8;
typedef __attribute__((ext_vector_type(4)))  int i32x4;

#define T_LEN   2048
#define D_MODEL 2048
#define HQ      16
#define HKV     4
#define HD      128
#define SINK    4
#define KV_LEN  (T_LEN + SINK)   // 2052 valid rows
#define KVP     2080             // padded row count
#define BT      4096             // B*T

#ifndef __has_builtin
#define __has_builtin(x) 0
#endif

#if __has_builtin(__builtin_amdgcn_tensor_load_to_lds)
#define USE_TDM 1
#else
#define USE_TDM 0
#endif

#if __has_builtin(__builtin_amdgcn_s_wait_tensorcnt)
#define WAIT_TENSORCNT0() __builtin_amdgcn_s_wait_tensorcnt(0)
#else
#define WAIT_TENSORCNT0() asm volatile("s_wait_tensorcnt 0x0" ::: "memory")
#endif

// ---------------------------------------------------------------------------
// WMMA helpers (CDNA5 16x16x32 bf16 -> f32)
// ---------------------------------------------------------------------------
__device__ inline v8f wmma_bf16(v16bf a, v16bf b, v8f c) {
  return __builtin_amdgcn_wmma_f32_16x16x32_bf16(
      /*neg_a=*/false, a, /*neg_b=*/false, b,
      /*c_mod=*/(short)0, c, /*reuse_a=*/false, /*reuse_b=*/false);
}

// A-matrix 16x32 bf16 fragment from a row-major (16 x ld) tile.
__device__ inline v16bf frag_a16(const __bf16* base, int ld) {
  const int lane = threadIdx.x & 31;
  const int row  = lane & 15;
  const int koff = (lane >> 4) * 8;
  const __bf16* p0 = base + row * ld + koff;
  bf16x8 lo = *(const bf16x8*)p0;
  bf16x8 hi = *(const bf16x8*)(p0 + 16);
  v16bf r;
#pragma unroll
  for (int e = 0; e < 8; ++e) { r[e] = lo[e]; r[e + 8] = hi[e]; }
  return r;
}

// B-matrix 32x16 bf16 fragment from an (N x K) row-major tile (B^T storage).
__device__ inline v16bf frag_b16(const __bf16* base, int ld) {
  const int lane = threadIdx.x & 31;
  const int col  = lane & 15;
  const int koff = (lane >> 4) * 8;
  const __bf16* p = base + col * ld + koff;
  bf16x8 lo = *(const bf16x8*)p;
  bf16x8 hi = *(const bf16x8*)(p + 16);
  v16bf r;
#pragma unroll
  for (int e = 0; e < 8; ++e) { r[e] = lo[e]; r[e + 8] = hi[e]; }
  return r;
}

#if USE_TDM
// LDS byte offset of a __shared__ pointer (AS3 ptrtoint)
__device__ inline unsigned lds_addr_of(const void* p) {
  return (unsigned)(unsigned long long)(const __attribute__((address_space(3))) void*)p;
}

// Build D# groups and issue TENSOR_LOAD_TO_LDS for a 2-D bf16 tile:
// width elems per row, height rows, global row stride (elems). LDS dest packed
// row-major (height x width). OOB (x >= tdim0) returns zero.
__device__ inline void tdm_load_2d(const void* gsrc, unsigned ldsoff,
                                   unsigned tdim0, unsigned tdim1,
                                   unsigned tile0, unsigned tile1,
                                   unsigned long long stride0) {
  unsigned long long ga = (unsigned long long)gsrc;
  u32x4 g0;
  g0[0] = 1u;                                            // count=1 (valid), user mode
  g0[1] = ldsoff;                                        // lds_addr
  g0[2] = (unsigned)(ga & 0xFFFFFFFFull);                // global_addr lo
  g0[3] = (unsigned)((ga >> 32) & 0x1FFFFFFull) | (2u << 30);  // addr hi | type=2
  i32x8 g1;
  g1[0] = (int)(1u << 16);                               // data_size=1 -> 2 bytes
  g1[1] = (int)((tdim0 & 0xFFFFu) << 16);                // tensor_dim0 lo16 @ [63:48]
  g1[2] = (int)(((tdim0 >> 16) & 0xFFFFu) | ((tdim1 & 0xFFFFu) << 16)); // td0 hi | td1 lo
  g1[3] = (int)(((tdim1 >> 16) & 0xFFFFu) | ((tile0 & 0xFFFFu) << 16)); // td1 hi | tile_dim0
  g1[4] = (int)(tile1 & 0xFFFFu);                        // tile_dim1 | tile_dim2=0
  g1[5] = (int)(unsigned)(stride0 & 0xFFFFFFFFull);      // tensor_dim0_stride lo32
  g1[6] = (int)(unsigned)((stride0 >> 32) & 0xFFFFull);  // stride hi16 | td1_stride lo=0
  g1[7] = 0;
  i32x4 z4; z4[0] = 0; z4[1] = 0; z4[2] = 0; z4[3] = 0;
  i32x8 z8;
#pragma unroll
  for (int i = 0; i < 8; ++i) z8[i] = 0;
  // 6-arg form (clang-23 / therock lane): groups 0..3 + extra group + cpol
  __builtin_amdgcn_tensor_load_to_lds(g0, g1, z4, z4, z8, 0);
}
#endif

// ---------------------------------------------------------------------------
// Elementwise f32 -> bf16 (vectorized, grid-stride over float4 units)
// ---------------------------------------------------------------------------
__global__ __launch_bounds__(256) void cvt_f32_bf16(
    const float* __restrict__ s, __bf16* __restrict__ d, int n4) {
  int i = blockIdx.x * 256 + threadIdx.x;
  int stride = gridDim.x * 256;
  for (; i < n4; i += stride) {
    float4 f = ((const float4*)s)[i];
    bf16x4 o;
    o[0] = (__bf16)f.x; o[1] = (__bf16)f.y; o[2] = (__bf16)f.z; o[3] = (__bf16)f.w;
    ((bf16x4*)d)[i] = o;
  }
}

// ---------------------------------------------------------------------------
// C = A * B^T (A: MxK bf16, B: NxK bf16, C: MxN f32), bf16 WMMA compute.
// Block tile 128x128, 8 waves, wave tile 32x64. Tiles staged by TDM (wave 0)
// when available, else plain loads. M%128==0, N%128==0, K%32==0.
// ---------------------------------------------------------------------------
__global__ __launch_bounds__(256) void gemm_bf16(
    const __bf16* __restrict__ A, const __bf16* __restrict__ B,
    float* __restrict__ C, int M, int N, int K) {
  __shared__ __bf16 As[128 * 32];
  __shared__ __bf16 Bs[128 * 32];
  const int m0 = blockIdx.y * 128;
  const int n0 = blockIdx.x * 128;
  const int t = threadIdx.x;
  const int w = t >> 5, lane = t & 31;
  const int half = lane >> 4, ln = lane & 15;
  const int wm = (w >> 1) * 32, wn = (w & 1) * 64;

  v8f acc[2][4];
#pragma unroll
  for (int i = 0; i < 2; ++i)
#pragma unroll
    for (int j = 0; j < 4; ++j)
#pragma unroll
      for (int e = 0; e < 8; ++e) acc[i][j][e] = 0.f;

  for (int k0 = 0; k0 < K; k0 += 32) {
    __syncthreads();
#if USE_TDM
    if (t < 32) {
      tdm_load_2d(A + (size_t)m0 * K + k0, lds_addr_of(As), 32, 128, 32, 128,
                  (unsigned long long)K);
      tdm_load_2d(B + (size_t)n0 * K + k0, lds_addr_of(Bs), 32, 128, 32, 128,
                  (unsigned long long)K);
      WAIT_TENSORCNT0();
    }
#else
#pragma unroll
    for (int u = 0; u < 2; ++u) {
      int idx = t + 256 * u;
      int r = idx >> 2, c = (idx & 3) * 8;
      *(bf16x8*)&As[r * 32 + c] = *(const bf16x8*)(A + (size_t)(m0 + r) * K + k0 + c);
      *(bf16x8*)&Bs[r * 32 + c] = *(const bf16x8*)(B + (size_t)(n0 + r) * K + k0 + c);
    }
#endif
    __syncthreads();

    v16bf af[2], bfr[4];
#pragma unroll
    for (int i = 0; i < 2; ++i) af[i] = frag_a16(As + (wm + i * 16) * 32, 32);
#pragma unroll
    for (int j = 0; j < 4; ++j) bfr[j] = frag_b16(Bs + (wn + j * 16) * 32, 32);
#pragma unroll
    for (int i = 0; i < 2; ++i)
#pragma unroll
      for (int j = 0; j < 4; ++j) acc[i][j] = wmma_bf16(af[i], bfr[j], acc[i][j]);
  }

#pragma unroll
  for (int i = 0; i < 2; ++i)
#pragma unroll
    for (int j = 0; j < 4; ++j)
#pragma unroll
      for (int r = 0; r < 8; ++r) {
        int row = m0 + wm + i * 16 + r + 8 * half;
        int col = n0 + wn + j * 16 + ln;
        C[(size_t)row * N + col] = acc[i][j][r];
      }
}

// ---------------------------------------------------------------------------
// RMSNorm + RoPE + scatter to bf16 Q/K/V caches. One block per (b,t).
// qkv row layout: [q 2048 | k 512 | v 512] (f32).
// ---------------------------------------------------------------------------
__global__ __launch_bounds__(256) void prepare_qkv(
    const float* __restrict__ qkv, const float* __restrict__ qg,
    const float* __restrict__ kgam, const int* __restrict__ pos_arr,
    __bf16* __restrict__ Qb, __bf16* __restrict__ Kb, __bf16* __restrict__ Vb) {
  const int bt = blockIdx.x;
  const int b = bt / T_LEN, tt = bt % T_LEN;
  const float* row = qkv + (size_t)bt * 3072;
  const int tid = threadIdx.x, w = tid >> 5, lane = tid & 31;

  float sq = 0.f, sk = 0.f;
#pragma unroll
  for (int i = 0; i < 8; ++i) { float v = row[tid + 256 * i]; sq += v * v; }
#pragma unroll
  for (int i = 0; i < 2; ++i) { float v = row[2048 + tid + 256 * i]; sk += v * v; }
#pragma unroll
  for (int off = 16; off >= 1; off >>= 1) {
    sq += __shfl_xor(sq, off, 32);
    sk += __shfl_xor(sk, off, 32);
  }
  __shared__ float shq[8], shk[8];
  if (lane == 0) { shq[w] = sq; shk[w] = sk; }
  __syncthreads();
  float tq = 0.f, tk = 0.f;
#pragma unroll
  for (int i = 0; i < 8; ++i) { tq += shq[i]; tk += shk[i]; }
  const float invq = rsqrtf(tq / 2048.f + 1e-5f);
  const float invk = rsqrtf(tk / 512.f + 1e-5f);
  const float pos = (float)pos_arr[tt];

#pragma unroll
  for (int u = 0; u < 4; ++u) {     // Q: 1024 rope pairs
    int p = tid + 256 * u;
    int h = p >> 6, i = p & 63;
    float fr = __powf(10000.f, -(float)i / 64.f);
    float ang = pos * fr, c = __cosf(ang), s = __sinf(ang);
    int base = h * 128;
    float x1 = row[base + i]      * invq * qg[base + i];
    float x2 = row[base + 64 + i] * invq * qg[base + 64 + i];
    __bf16* dst = Qb + ((size_t)(b * HQ + h) * T_LEN + tt) * 128;
    dst[i]      = (__bf16)(x1 * c - x2 * s);
    dst[64 + i] = (__bf16)(x1 * s + x2 * c);
  }
  {                                 // K: 256 rope pairs
    int p = tid, h = p >> 6, i = p & 63;
    float fr = __powf(10000.f, -(float)i / 64.f);
    float ang = pos * fr, c = __cosf(ang), s = __sinf(ang);
    const float* krow = row + 2048;
    int base = h * 128;
    float x1 = krow[base + i]      * invk * kgam[base + i];
    float x2 = krow[base + 64 + i] * invk * kgam[base + 64 + i];
    __bf16* dst = Kb + ((size_t)(b * HKV + h) * KVP + SINK + tt) * 128;
    dst[i]      = (__bf16)(x1 * c - x2 * s);
    dst[64 + i] = (__bf16)(x1 * s + x2 * c);
  }
#pragma unroll
  for (int u = 0; u < 2; ++u) {     // V: 512 values
    int p = tid + 256 * u;
    int h = p >> 7, i = p & 127;
    Vb[((size_t)(b * HKV + h) * KVP + SINK + tt) * 128 + i] = (__bf16)row[2560 + p];
  }
}

__global__ void fill_sinks(const float* __restrict__ ks, const float* __restrict__ vs,
                           __bf16* __restrict__ Kb, __bf16* __restrict__ Vb) {
  int idx = blockIdx.x;               // 0..31 = b*16 + h*4 + s
  int b = idx >> 4, r = idx & 15, h = r >> 2, s = r & 3;
  int i = threadIdx.x;                // 0..127
  size_t dst = ((size_t)(b * HKV + h) * KVP + s) * 128 + i;
  size_t src = (size_t)(h * SINK + s) * 128 + i;
  Kb[dst] = (__bf16)ks[src];
  Vb[dst] = (__bf16)vs[src];
}

// ---------------------------------------------------------------------------
// Flash attention. Block = 128 query rows x (b,h). 8 waves x 16 rows.
// K tile staged by TDM (1-D copy; tensor_dim0 = remaining valid elems gives
// hardware zero-fill past KV_LEN). V staged manually with transpose.
// ---------------------------------------------------------------------------
__global__ __launch_bounds__(256) void flash_attn(
    const __bf16* __restrict__ Qb, const __bf16* __restrict__ Kb,
    const __bf16* __restrict__ Vb, __bf16* __restrict__ yb) {
  __shared__ __bf16 Ksm[32 * 128];    // [key][d]
  __shared__ __bf16 Vst[128 * 48];    // [d][key] transposed, stride 48
  __shared__ __bf16 Ps[8 * 16 * 32];  // per-wave P scratch

  const int bh = blockIdx.y, b = bh >> 4, h = bh & 15, kvh = h >> 2;
  const int qbase = blockIdx.x * 128;
  const int tid = threadIdx.x, w = tid >> 5, lane = tid & 31;
  const int half = lane >> 4, ln = lane & 15;
  const int qrow0 = qbase + w * 16;

  const __bf16* Qtile = Qb + ((size_t)bh * T_LEN + qrow0) * 128;
  v16bf aq[4];
#pragma unroll
  for (int c = 0; c < 4; ++c) aq[c] = frag_a16(Qtile + c * 32, 128);

  const __bf16* Kbase = Kb + (size_t)(b * HKV + kvh) * KVP * 128;
  const __bf16* Vbase = Vb + (size_t)(b * HKV + kvh) * KVP * 128;

  float m_r[8], l_r[8];
  v8f o[8];
#pragma unroll
  for (int r = 0; r < 8; ++r) { m_r[r] = -1e30f; l_r[r] = 0.f; }
#pragma unroll
  for (int j = 0; j < 8; ++j)
#pragma unroll
    for (int e = 0; e < 8; ++e) o[j][e] = 0.f;

  const float scale = 0.08838834764831845f;  // 1/sqrt(128)
  const int iters = (SINK + qbase + 128 + 31) >> 5;

  for (int it = 0; it < iters; ++it) {
    const int kt = it * 32;
    __syncthreads();
#if USE_TDM
    if (tid < 32) {
      unsigned valid = (unsigned)((KV_LEN - kt) * 128);   // kt < KV_LEN always
      tdm_load_2d(Kbase + (size_t)kt * 128, lds_addr_of(Ksm),
                  valid, 1, 32 * 128, 1, (unsigned long long)valid);
    }
#endif
#pragma unroll
    for (int u2 = 0; u2 < 2; ++u2) {
      int u = tid + 256 * u2;
      int kr = u >> 4, dc = (u & 15) * 8;
      int kgl = kt + kr;
      bf16x8 vv;
#pragma unroll
      for (int e = 0; e < 8; ++e) vv[e] = (__bf16)0.f;
#if !USE_TDM
      bf16x8 kv;
#pragma unroll
      for (int e = 0; e < 8; ++e) kv[e] = (__bf16)0.f;
      if (kgl < KV_LEN) kv = *(const bf16x8*)(Kbase + (size_t)kgl * 128 + dc);
      *(bf16x8*)&Ksm[kr * 128 + dc] = kv;
#endif
      if (kgl < KV_LEN) vv = *(const bf16x8*)(Vbase + (size_t)kgl * 128 + dc);
#pragma unroll
      for (int e = 0; e < 8; ++e) Vst[(dc + e) * 48 + kr] = vv[e];
    }
#if USE_TDM
    if (tid < 32) WAIT_TENSORCNT0();
#endif
    __syncthreads();

    // scores: Q(16x128) x K(32x128)^T -> two 16x16 tiles
    v8f s0, s1;
#pragma unroll
    for (int e = 0; e < 8; ++e) { s0[e] = 0.f; s1[e] = 0.f; }
#pragma unroll
    for (int c = 0; c < 4; ++c) {
      v16bf bk0 = frag_b16(Ksm + c * 32, 128);
      v16bf bk1 = frag_b16(Ksm + 16 * 128 + c * 32, 128);
      s0 = wmma_bf16(aq[c], bk0, s0);
      s1 = wmma_bf16(aq[c], bk1, s1);
    }

    // online softmax per row
    __bf16* pw = Ps + w * 16 * 32;
#pragma unroll
    for (int r = 0; r < 8; ++r) {
      int qpos = qrow0 + r + 8 * half;
      int kg0 = kt + ln, kg1 = kt + 16 + ln;
      float x0 = (kg0 <= qpos + SINK) ? s0[r] * scale : -1e30f;
      float x1 = (kg1 <= qpos + SINK) ? s1[r] * scale : -1e30f;
      float mx = fmaxf(x0, x1);
#pragma unroll
      for (int off = 1; off < 16; off <<= 1) mx = fmaxf(mx, __shfl_xor(mx, off, 32));
      float mnew = fmaxf(m_r[r], mx);
      float p0 = __expf(x0 - mnew);
      float p1 = __expf(x1 - mnew);
      float rs = p0 + p1;
#pragma unroll
      for (int off = 1; off < 16; off <<= 1) rs += __shfl_xor(rs, off, 32);
      float corr = __expf(m_r[r] - mnew);
      l_r[r] = l_r[r] * corr + rs;
      m_r[r] = mnew;
#pragma unroll
      for (int j = 0; j < 8; ++j) o[j][r] *= corr;
      int prow = r + 8 * half;
      pw[prow * 32 + ln]      = (__bf16)p0;
      pw[prow * 32 + 16 + ln] = (__bf16)p1;
    }

    // O += P(16x32) x V(32x128)
    v16bf pf = frag_a16(pw, 32);
#pragma unroll
    for (int j = 0; j < 8; ++j) {
      v16bf vf = frag_b16(Vst + (j * 16) * 48, 48);
      o[j] = wmma_bf16(pf, vf, o[j]);
    }
  }

  // epilogue: divide by l, write y (B,T,Hq*d) bf16 for the output GEMM
#pragma unroll
  for (int j = 0; j < 8; ++j)
#pragma unroll
    for (int r = 0; r < 8; ++r) {
      int trow = qrow0 + r + 8 * half;
      int col = h * 128 + j * 16 + ln;
      yb[(size_t)(b * T_LEN + trow) * D_MODEL + col] = (__bf16)(o[j][r] / l_r[r]);
    }
}

// ---------------------------------------------------------------------------
extern "C" void kernel_launch(void* const* d_in, const int* in_sizes, int n_in,
                              void* d_out, int out_size, void* d_ws, size_t ws_size,
                              hipStream_t stream) {
  const float* x   = (const float*)d_in[0];
  const float* Wc  = (const float*)d_in[1];
  const float* qg  = (const float*)d_in[2];
  const float* kgm = (const float*)d_in[3];
  const float* ksk = (const float*)d_in[4];
  const float* vsk = (const float*)d_in[5];
  const float* Wp  = (const float*)d_in[6];
  const int*   pos = (const int*)d_in[7];

  char* ws = (char*)d_ws;
  size_t off = 0;
  float*  qkv = (float*)(ws + off);  off += (size_t)BT * 3072 * sizeof(float);       // 48 MB
  __bf16* Qb  = (__bf16*)(ws + off); off += (size_t)2 * HQ * T_LEN * 128 * 2;        // 16 MB
  __bf16* Kb  = (__bf16*)(ws + off); off += (size_t)2 * HKV * KVP * 128 * 2;         // ~4.3 MB
  __bf16* Vb  = (__bf16*)(ws + off); off += (size_t)2 * HKV * KVP * 128 * 2;         // ~4.3 MB
  __bf16* xb  = (__bf16*)(ws + off); off += (size_t)BT * D_MODEL * 2;                // 16 MB
  __bf16* Wcb = (__bf16*)(ws + off); off += (size_t)3072 * D_MODEL * 2;              // 12 MB
  __bf16* yb  = xb;   // xb dead after GEMM1
  __bf16* Wpb = Wcb;  // Wcb dead after GEMM1 (Wp converted after GEMM1)

  // 0) convert inputs to bf16
  cvt_f32_bf16<<<2048, 256, 0, stream>>>(x,  xb,  BT * D_MODEL / 4);
  cvt_f32_bf16<<<2048, 256, 0, stream>>>(Wc, Wcb, 3072 * D_MODEL / 4);
  // 1) qkv = x @ Wc^T
  gemm_bf16<<<dim3(3072 / 128, BT / 128), 256, 0, stream>>>(xb, Wcb, qkv, BT, 3072, D_MODEL);
  // 2) rmsnorm + rope + scatter
  prepare_qkv<<<BT, 256, 0, stream>>>(qkv, qg, kgm, pos, Qb, Kb, Vb);
  // 3) sink rows
  fill_sinks<<<2 * HKV * SINK, 128, 0, stream>>>(ksk, vsk, Kb, Vb);
  // 3b) convert Wp (into Wcb's space, now dead)
  cvt_f32_bf16<<<2048, 256, 0, stream>>>(Wp, Wpb, D_MODEL * D_MODEL / 4);
  // 4) attention -> yb (bf16, reuses xb space)
  flash_attn<<<dim3(T_LEN / 128, 2 * HQ), 256, 0, stream>>>(Qb, Kb, Vb, yb);
  // 5) out = y @ Wp^T
  gemm_bf16<<<dim3(D_MODEL / 128, BT / 128), 256, 0, stream>>>(yb, Wpb, (float*)d_out, BT, D_MODEL, D_MODEL);
}
